// MultiHeadAttentionBlock_23330262352161
// MI455X (gfx1250) — compile-verified, matrix operand order checked by reading, not run on hardware
//
#include <hip/hip_runtime.h>
#include <hip/hip_bf16.h>
#include <math.h>
#include <stdint.h>

typedef __bf16 bf16;
typedef __attribute__((ext_vector_type(16))) __bf16 v16bf;
typedef __attribute__((ext_vector_type(8)))  __bf16 v8bf;
typedef __attribute__((ext_vector_type(8)))  float  v8f;

union FragA { v16bf v; v8bf h[2]; };

#define BB 2
#define TT 2048
#define SS 2048
#define DM 1024
#define NH 16
#define HD 64

// ---------------------------------------------------------------------------
// CDNA5 async global->LDS copy (16B per lane), tracked by ASYNCcnt.
// VDST = VGPR holding 32-bit LDS byte address; VADDR = 64-bit global address.
// ---------------------------------------------------------------------------
__device__ __forceinline__ void async_ld16(const void* g, void* l) {
    unsigned lds_off = (unsigned)(uintptr_t)l;              // LDS aperture: low 32 bits
    unsigned long long ga = (unsigned long long)(uintptr_t)g;
    asm volatile("global_load_async_to_lds_b128 %0, %1, off"
                 :: "v"(lds_off), "v"(ga) : "memory");
}
__device__ __forceinline__ void wait_async0() {
    asm volatile("s_wait_asynccnt 0x0" ::: "memory");
}

// ---------------------------------------------------------------------------
// f32 -> bf16 conversion (grid-stride); sources are single-use -> NT loads
// ---------------------------------------------------------------------------
__global__ __launch_bounds__(256) void cvt_f32_bf16(const float* __restrict__ src,
                                                    bf16* __restrict__ dst, int n) {
    int i = blockIdx.x * blockDim.x + threadIdx.x;
    int stride = gridDim.x * blockDim.x;
    for (; i < n; i += stride) dst[i] = (bf16)__builtin_nontemporal_load(src + i);
}

// ---------------------------------------------------------------------------
// C[M,N] = X[M,K] * W[N,K]^T   (bf16 in, f32 out), 128x64 tile, 8 waves,
// double-buffered async-to-LDS staging, one barrier per K-step.
// ---------------------------------------------------------------------------
#define GLDK 40
#define XTILE (128 * GLDK)
#define WTILE (64 * GLDK)
__global__ __launch_bounds__(256) void gemm_xwT_bf16(const bf16* __restrict__ X,
                                                     const bf16* __restrict__ W,
                                                     float* __restrict__ Out,
                                                     int M, int N, int K) {
    __shared__ __align__(16) bf16 xs[2 * XTILE];
    __shared__ __align__(16) bf16 wt[2 * WTILE];
    const int tid   = threadIdx.x;
    const int lane  = tid & 31;
    const int wave  = tid >> 5;
    const int lhalf = lane >> 4;
    const int lmod  = lane & 15;
    const int m0 = blockIdx.y * 128;
    const int n0 = blockIdx.x * 64;

    v8f acc[4] = {};

    const int xrow = tid >> 1, xseg = tid & 1;   // 128 rows x 2 segs of 16
    const int wrow = tid >> 2, wseg = tid & 3;   // 64 rows x 4 segs of 8

    const bf16* xg = X + (size_t)(m0 + xrow) * K + xseg * 16;
    const bf16* wg = W + (size_t)(n0 + wrow) * K + wseg * 8;
    const int xloff = xrow * GLDK + xseg * 16;
    const int wloff = wrow * GLDK + wseg * 8;

    // prologue: stage K-step 0 into buffer 0
    async_ld16(xg,     xs + xloff);
    async_ld16(xg + 8, xs + xloff + 8);
    async_ld16(wg,     wt + wloff);

    for (int k0 = 0; k0 < K; k0 += 32) {
        const int cur = (k0 >> 5) & 1;
        wait_async0();              // tile issued last iteration is complete
        __syncthreads();            // ... for every wave; prev buffer reads done
        if (k0 + 32 < K) {          // issue next tile into other buffer (overlaps WMMA)
            const int nxt = 1 - cur;
            const bf16* xn = xg + k0 + 32;
            const bf16* wn = wg + k0 + 32;
            async_ld16(xn,     xs + nxt * XTILE + xloff);
            async_ld16(xn + 8, xs + nxt * XTILE + xloff + 8);
            async_ld16(wn,     wt + nxt * WTILE + wloff);
            __builtin_prefetch(xg + k0 + 64, 0, 3);
            __builtin_prefetch(wg + k0 + 64, 0, 3);
        }
        FragA a;
        const bf16* xp = xs + cur * XTILE + (wave * 16 + lmod) * GLDK + lhalf * 8;
        a.h[0] = *(const v8bf*)(xp);
        a.h[1] = *(const v8bf*)(xp + 16);
#pragma unroll
        for (int ct = 0; ct < 4; ++ct) {
            FragA bf;
            const bf16* wp = wt + cur * WTILE + (ct * 16 + lmod) * GLDK + lhalf * 8;
            bf.h[0] = *(const v8bf*)(wp);
            bf.h[1] = *(const v8bf*)(wp + 16);
            acc[ct] = __builtin_amdgcn_wmma_f32_16x16x32_bf16(
                false, a.v, false, bf.v, (short)0, acc[ct], false, false);
        }
    }
#pragma unroll
    for (int ct = 0; ct < 4; ++ct)
#pragma unroll
        for (int r = 0; r < 8; ++r) {
            int row = m0 + wave * 16 + r + lhalf * 8;
            int col = n0 + ct * 16 + lmod;
            Out[(size_t)row * N + col] = acc[ct][r];
        }
}

// ---------------------------------------------------------------------------
// L2-normalize per head (+ logit scale for q), repack to [B,H,T,64] bf16.
// One wave per head, one (b,t) row per block. grid.y: 0=q,1=k,2=v
// ---------------------------------------------------------------------------
__global__ __launch_bounds__(512) void normpack(const float* __restrict__ qp,
                                                const float* __restrict__ kp,
                                                const float* __restrict__ vp,
                                                const float* __restrict__ logit_scale,
                                                bf16* __restrict__ qn,
                                                bf16* __restrict__ kn,
                                                bf16* __restrict__ vn) {
    const int row    = blockIdx.x;        // b*T + t
    const int tensor = blockIdx.y;
    const int h    = threadIdx.x >> 5;
    const int lane = threadIdx.x & 31;
    const int e    = lane * 2;
    const float* src = tensor == 0 ? qp : (tensor == 1 ? kp : vp);
    bf16*        dst = tensor == 0 ? qn : (tensor == 1 ? kn : vn);
    const int b = row >> 11;
    const int t = row & (TT - 1);

    float x0 = src[(size_t)row * DM + h * HD + e];
    float x1 = src[(size_t)row * DM + h * HD + e + 1];
    float scale = 1.0f;
    if (tensor < 2) {
        float s = x0 * x0 + x1 * x1;
#pragma unroll
        for (int m = 1; m < 32; m <<= 1) s += __shfl_xor(s, m, 32);
        float nrm = sqrtf(s);
        scale = 1.0f / fmaxf(nrm, 1e-12f);
        if (tensor == 0)
            scale *= __expf(fminf(logit_scale[h], 4.6051702f)); // ln(100)
    }
    size_t o = (((size_t)b * NH + h) * TT + t) * HD + e;
    dst[o]     = (bf16)(x0 * scale);
    dst[o + 1] = (bf16)(x1 * scale);
}

// ---------------------------------------------------------------------------
// vn [BH,S,64] -> vt [BH,64,S]  (bf16 tiled transpose)
// ---------------------------------------------------------------------------
__global__ __launch_bounds__(256) void transpose64(const bf16* __restrict__ src,
                                                   bf16* __restrict__ dst) {
    __shared__ __align__(16) bf16 tile[64 * 72];
    const int bh = blockIdx.y;
    const int s0 = blockIdx.x * 64;
    const int tid = threadIdx.x;
    const int r = tid >> 2, seg = tid & 3;
    const bf16* sp = src + ((size_t)bh * SS + s0 + r) * HD + seg * 16;
    async_ld16(sp,     tile + r * 72 + seg * 16);
    async_ld16(sp + 8, tile + r * 72 + seg * 16 + 8);
    wait_async0();
    __syncthreads();
    const int d = tid >> 2;
    union { v8bf v; bf16 e[8]; } u0, u1;
#pragma unroll
    for (int i = 0; i < 8; ++i) {
        u0.e[i] = tile[(seg * 16 + i) * 72 + d];
        u1.e[i] = tile[(seg * 16 + 8 + i) * 72 + d];
    }
    bf16* dp = dst + ((size_t)bh * HD + d) * SS + s0 + seg * 16;
    *(v8bf*)(dp)     = u0.v;
    *(v8bf*)(dp + 8) = u1.v;
}

// ---------------------------------------------------------------------------
// Flash attention: one (b,h) x 128 query rows per block; 8 waves x 16 rows.
// Double-buffered async K/V^T staging, one barrier per j-step; bias loads
// issued before the async wait so they fly under the DMA + QK^T WMMAs.
// qn,kn: [B,H,T|S,64] bf16; vt: [B,H,64,S] bf16; bias: [T,S,H] f32.
// ---------------------------------------------------------------------------
#define ALDP 72
#define KTILE (64 * ALDP)
__global__ __launch_bounds__(256) void flash_attn(const bf16* __restrict__ qn,
                                                  const bf16* __restrict__ kn,
                                                  const bf16* __restrict__ vt,
                                                  const float* __restrict__ bias,
                                                  bf16* __restrict__ outbf) {
    __shared__ __align__(16) bf16 ks[2 * KTILE];
    __shared__ __align__(16) bf16 vs[2 * KTILE];
    __shared__ __align__(16) bf16 ps[8 * 16 * ALDP];
    const int bh = blockIdx.x;
    const int b = bh >> 4, h = bh & 15;
    const int t0 = blockIdx.y * 128;
    const int tid = threadIdx.x, lane = tid & 31, wave = tid >> 5;
    const int lhalf = lane >> 4, lmod = lane & 15;

    // Q fragments (rows t0 + 16*wave + m), K-dim = head dim 64 -> two frags
    FragA qa0, qa1;
    {
        const bf16* p0 = qn + (((size_t)bh * TT) + t0 + wave * 16 + lmod) * HD + lhalf * 8;
        qa0.h[0] = *(const v8bf*)(p0);
        qa0.h[1] = *(const v8bf*)(p0 + 16);
        qa1.h[0] = *(const v8bf*)(p0 + 32);
        qa1.h[1] = *(const v8bf*)(p0 + 48);
    }
    v8f oacc[4] = {};
    float mrow[8], lrow[8];
#pragma unroll
    for (int r = 0; r < 8; ++r) { mrow[r] = -1e30f; lrow[r] = 0.0f; }

    const int krow = tid >> 2, kseg = tid & 3;
    bf16* pw = ps + wave * 16 * ALDP;
    const int trow = t0 + wave * 16 + lhalf * 8;

    const bf16* kg = kn + (((size_t)bh * SS) + krow) * HD + kseg * 16;
    const bf16* vg = vt + (((size_t)bh * HD) + krow) * SS + kseg * 16;
    const int kloff = krow * ALDP + kseg * 16;

    // prologue: stage j=0 tiles into buffer 0
    async_ld16(kg,     ks + kloff);
    async_ld16(kg + 8, ks + kloff + 8);
    async_ld16(vg,     vs + kloff);
    async_ld16(vg + 8, vs + kloff + 8);

    const int nj = SS / 64;
    for (int jj = 0; jj < nj; ++jj) {
        const int j = jj * 64;
        const int cur = jj & 1;
        // bias loads: independent of LDS -> issue before the async drain
        float bb[4][8];
#pragma unroll
        for (int ct = 0; ct < 4; ++ct)
#pragma unroll
            for (int r = 0; r < 8; ++r)
                bb[ct][r] = bias[((size_t)(trow + r) * SS + j + ct * 16 + lmod) * NH + h];

        wait_async0();              // K/V tile for this j is in LDS
        __syncthreads();            // ... for all waves; old buffer reads done
        if (jj + 1 < nj) {          // issue next tile into other buffer
            const int nxt = 1 - cur;
            const bf16* kgn = kg + (size_t)(j + 64) * HD;
            const bf16* vgn = vg + (j + 64);
            async_ld16(kgn,     ks + nxt * KTILE + kloff);
            async_ld16(kgn + 8, ks + nxt * KTILE + kloff + 8);
            async_ld16(vgn,     vs + nxt * KTILE + kloff);
            async_ld16(vgn + 8, vs + nxt * KTILE + kloff + 8);
        }
        const bf16* ksc = ks + cur * KTILE;
        const bf16* vsc = vs + cur * KTILE;

        // S = Q K^T
        v8f sacc[4] = {};
#pragma unroll
        for (int ct = 0; ct < 4; ++ct) {
            FragA kb0, kb1;
            const bf16* kp0 = ksc + (ct * 16 + lmod) * ALDP + lhalf * 8;
            kb0.h[0] = *(const v8bf*)(kp0);
            kb0.h[1] = *(const v8bf*)(kp0 + 16);
            kb1.h[0] = *(const v8bf*)(kp0 + 32);
            kb1.h[1] = *(const v8bf*)(kp0 + 48);
            sacc[ct] = __builtin_amdgcn_wmma_f32_16x16x32_bf16(
                false, qa0.v, false, kb0.v, (short)0, sacc[ct], false, false);
            sacc[ct] = __builtin_amdgcn_wmma_f32_16x16x32_bf16(
                false, qa1.v, false, kb1.v, (short)0, sacc[ct], false, false);
        }
        // + bias, online softmax
#pragma unroll
        for (int ct = 0; ct < 4; ++ct)
#pragma unroll
            for (int r = 0; r < 8; ++r) sacc[ct][r] += bb[ct][r];
#pragma unroll
        for (int r = 0; r < 8; ++r) {
            float mx = fmaxf(fmaxf(sacc[0][r], sacc[1][r]),
                             fmaxf(sacc[2][r], sacc[3][r]));
            mx = fmaxf(mx, __shfl_xor(mx, 1, 32));
            mx = fmaxf(mx, __shfl_xor(mx, 2, 32));
            mx = fmaxf(mx, __shfl_xor(mx, 4, 32));
            mx = fmaxf(mx, __shfl_xor(mx, 8, 32));
            float mnew  = fmaxf(mrow[r], mx);
            float alpha = __expf(mrow[r] - mnew);
            float rs = 0.0f;
#pragma unroll
            for (int ct = 0; ct < 4; ++ct) {
                float p = __expf(sacc[ct][r] - mnew);
                sacc[ct][r] = p;
                rs += p;
            }
            rs += __shfl_xor(rs, 1, 32);
            rs += __shfl_xor(rs, 2, 32);
            rs += __shfl_xor(rs, 4, 32);
            rs += __shfl_xor(rs, 8, 32);
            lrow[r] = lrow[r] * alpha + rs;
            mrow[r] = mnew;
#pragma unroll
            for (int ct = 0; ct < 4; ++ct) oacc[ct][r] *= alpha;
        }
        // P -> per-wave LDS pad (C-layout -> A-layout); wave-private region,
        // cross-lane visibility needs only the compiler's s_wait_dscnt.
#pragma unroll
        for (int ct = 0; ct < 4; ++ct)
#pragma unroll
            for (int r = 0; r < 8; ++r)
                pw[(r + lhalf * 8) * ALDP + ct * 16 + lmod] = (bf16)sacc[ct][r];

        FragA pa0, pa1;
        const bf16* pp = pw + lmod * ALDP + lhalf * 8;
        pa0.h[0] = *(const v8bf*)(pp);
        pa0.h[1] = *(const v8bf*)(pp + 16);
        pa1.h[0] = *(const v8bf*)(pp + 32);
        pa1.h[1] = *(const v8bf*)(pp + 48);
        // O += P * V
#pragma unroll
        for (int ct = 0; ct < 4; ++ct) {
            FragA vb0, vb1;
            const bf16* vp0 = vsc + (ct * 16 + lmod) * ALDP + lhalf * 8;
            vb0.h[0] = *(const v8bf*)(vp0);
            vb0.h[1] = *(const v8bf*)(vp0 + 16);
            vb1.h[0] = *(const v8bf*)(vp0 + 32);
            vb1.h[1] = *(const v8bf*)(vp0 + 48);
            oacc[ct] = __builtin_amdgcn_wmma_f32_16x16x32_bf16(
                false, pa0.v, false, vb0.v, (short)0, oacc[ct], false, false);
            oacc[ct] = __builtin_amdgcn_wmma_f32_16x16x32_bf16(
                false, pa1.v, false, vb1.v, (short)0, oacc[ct], false, false);
        }
    }
    // epilogue: O / l  -> [B,T,1024] bf16 (input to out-projection GEMM)
#pragma unroll
    for (int ct = 0; ct < 4; ++ct)
#pragma unroll
        for (int r = 0; r < 8; ++r) {
            int t = trow + r;
            int d = h * HD + ct * 16 + lmod;
            float val = oacc[ct][r] / fmaxf(lrow[r], 1e-30f);
            outbf[((size_t)b * TT + t) * DM + d] = (bf16)val;
        }
}

// ---------------------------------------------------------------------------
extern "C" void kernel_launch(void* const* d_in, const int* in_sizes, int n_in,
                              void* d_out, int out_size, void* d_ws, size_t ws_size,
                              hipStream_t stream) {
    const float* q    = (const float*)d_in[0];
    const float* k    = (const float*)d_in[1];
    const float* v    = (const float*)d_in[2];
    const float* bias = (const float*)d_in[3];
    const float* Wqkv = (const float*)d_in[4];
    const float* Wout = (const float*)d_in[5];
    const float* lsc  = (const float*)d_in[6];

    char* wsb = (char*)d_ws;
    size_t off = 0;
    auto alloc = [&](size_t bytes) -> void* {
        void* p = wsb + off;
        off = (off + bytes + 255) & ~(size_t)255;
        return p;
    };
    const int NX = BB * TT * DM;            // 4,194,304 elements
    bf16* qbf    = (bf16*)alloc((size_t)NX * 2);        // later reused as qn
    bf16* kbf    = (bf16*)alloc((size_t)NX * 2);        // later reused as kn
    bf16* vbf    = (bf16*)alloc((size_t)NX * 2);        // later reused as vn
    bf16* wqkvbf = (bf16*)alloc((size_t)DM * DM * 2);
    bf16* woutbf = (bf16*)alloc((size_t)DM * DM * 2);
    float* qp    = (float*)alloc((size_t)NX * 4);       // later reused as attnbf
    float* kp    = (float*)alloc((size_t)NX * 4);       // later reused as vt
    float* vp    = (float*)alloc((size_t)NX * 4);
    bf16* qn = qbf, *kn = kbf, *vn = vbf;
    bf16* vt     = (bf16*)kp;                           // kp dead after normpack
    bf16* attnbf = (bf16*)qp;                           // qp dead after normpack

    // 1) f32 -> bf16
    cvt_f32_bf16<<<2048, 256, 0, stream>>>(q, qbf, NX);
    cvt_f32_bf16<<<2048, 256, 0, stream>>>(k, kbf, NX);
    cvt_f32_bf16<<<2048, 256, 0, stream>>>(v, vbf, NX);
    cvt_f32_bf16<<<1024, 256, 0, stream>>>(Wqkv, wqkvbf, DM * DM);
    cvt_f32_bf16<<<1024, 256, 0, stream>>>(Wout, woutbf, DM * DM);

    // 2) QKV projections: [4096,1024] x Wqkv^T
    dim3 gg(DM / 64, (BB * TT) / 128);
    gemm_xwT_bf16<<<gg, 256, 0, stream>>>(qbf, wqkvbf, qp, BB * TT, DM, DM);
    gemm_xwT_bf16<<<gg, 256, 0, stream>>>(kbf, wqkvbf, kp, BB * TT, DM, DM);
    gemm_xwT_bf16<<<gg, 256, 0, stream>>>(vbf, wqkvbf, vp, BB * TT, DM, DM);

    // 3) per-head L2 norm (+logit scale), repack heads
    normpack<<<dim3(BB * TT, 3), 512, 0, stream>>>(qp, kp, vp, lsc, qn, kn, vn);

    // 3b) V -> V^T per (b,h)
    transpose64<<<dim3(SS / 64, BB * NH), 256, 0, stream>>>(vn, vt);

    // 4) flash attention (bh fast dim -> bias L2 reuse across heads/batch)
    flash_attn<<<dim3(BB * NH, TT / 128), 256, 0, stream>>>(qn, kn, vt, bias, attnbf);

    // 5) out projection -> f32 d_out
    gemm_xwT_bf16<<<gg, 256, 0, stream>>>(attnbf, woutbf, (float*)d_out,
                                          BB * TT, DM, DM);
}
